// LinearAttention_70626442216109
// MI455X (gfx1250) — matrix operand order repaired and checked by reading, not compiled, not run on hardware
//
#include <hip/hip_runtime.h>

// ---------------------------------------------------------------------------
// Linear attention for MI455X (gfx1250, wave32, WMMA + async LDS staging).
// Pipeline:
//   1) x  (f32) -> X16 (f16)
//   2) W_qkv, W_out -> transposed f16 (B-operand layout: [n][k], k contiguous)
//   3) QKV16[BN][3072] = X16 x WqkvT            (WMMA f16 -> f32 -> f16)
//   4) per (b,h): softmax_N(k) -> Kt[d][n], v -> Vt[e][n]  (LDS transpose)
//   5) per (b,h): CTX16[e][d] = Vt x Kt^T       (= context^T, WMMA)
//   6) per (b,h): ATTN16[n][h*64+e] = q x ctx   (WMMA, q read in-place)
//   7) d_out = ATTN16 x WoutT + b_out           (WMMA, f32 out)
// GEMM uses GLOBAL_LOAD_ASYNC_TO_LDS_B128 double-buffered tiles (ASYNCcnt),
// ds_load_b128 fragment reads, v_wmma_f32_16x16x32_f16 compute.
// ---------------------------------------------------------------------------

typedef _Float16 half_t;
typedef __attribute__((ext_vector_type(16))) _Float16 v16h;
typedef __attribute__((ext_vector_type(8)))  float    v8f;
typedef __attribute__((ext_vector_type(4)))  int      v4i;

static constexpr int  Bb = 4, Nn = 4096, Dd = 1024, Hh = 16, HD = 64;
static constexpr long BN = (long)Bb * Nn;      // 16384
static constexpr int  D3 = 3 * Dd;             // 3072

static constexpr int KSTEP = 32;               // K per stage
static constexpr int LROW  = 40;               // padded halves per LDS row (80B)

union FragU { v16h v; uint4 q[2]; };

__device__ __forceinline__ v16h load_frag2(const half_t* c0, const half_t* c1) {
  FragU f;
  f.q[0] = *(const uint4*)c0;
  f.q[1] = *(const uint4*)c1;
  return f.v;
}

__device__ __forceinline__ int imin(int a, int b) { return a < b ? a : b; }

// -------- CDNA5 async global->LDS copy (ASYNCcnt-tracked) -------------------
typedef __attribute__((address_space(1))) v4i* gptr_v4i;   // global int4*
typedef __attribute__((address_space(3))) v4i* lptr_v4i;   // LDS int4*

__device__ __forceinline__ void async_copy16(const void* g, void* l) {
#if __has_builtin(__builtin_amdgcn_global_load_async_to_lds_b128)
  __builtin_amdgcn_global_load_async_to_lds_b128(
      (gptr_v4i)(unsigned long long)g,
      (lptr_v4i)(unsigned)(unsigned long long)l, 0, 0);
#else
  unsigned loff = (unsigned)(unsigned long long)l;
  asm volatile("global_load_async_to_lds_b128 %0, %1, off"
               :: "v"(loff), "v"(g) : "memory");
#endif
}

__device__ __forceinline__ void wait_async0() {
#if __has_builtin(__builtin_amdgcn_s_wait_asynccnt)
  __builtin_amdgcn_s_wait_asynccnt(0);
#else
  asm volatile("s_wait_asynccnt 0x0" ::: "memory");
#endif
}

__device__ __forceinline__ void wait_async3() {
#if __has_builtin(__builtin_amdgcn_s_wait_asynccnt)
  __builtin_amdgcn_s_wait_asynccnt(3);
#else
  asm volatile("s_wait_asynccnt 0x3" ::: "memory");
#endif
}

// ---------------------------------------------------------------------------
// Generic batched WMMA GEMM:  C[m][n] = sum_k A[m][k] * Bt[n][k]  (+bias)
// Block: 256 threads = 8 waves; block tile 128(M) x 64(N); wave tile 32x32.
// A/B tiles double-buffered in LDS via async copies; K-step 32.
// Batch z decomposes as (zb, zh) = (z / H2, z % H2) with per-dim strides.
// Output: f16 to Ch if Ch != nullptr, else f32 (+bias) to Cf.
// ---------------------------------------------------------------------------
__device__ __forceinline__ void store_tile(v8f acc, int row, int col,
                                           int M, int N, long ldc,
                                           float* Cf, half_t* Ch,
                                           const float* bias) {
  if (col >= N) return;
  float bv = bias ? bias[col] : 0.0f;
#pragma unroll
  for (int r = 0; r < 8; ++r) {
    int rr = row + r;
    if (rr >= M) return;
    if (Ch) Ch[(long)rr * ldc + col] = (half_t)acc[r];
    else    Cf[(long)rr * ldc + col] = acc[r] + bv;
  }
}

__global__ __launch_bounds__(256)
void gemm_wmma_f16(const half_t* __restrict__ A, long lda, long sAb, long sAh,
                   const half_t* __restrict__ Bt, long ldb, long sBb, long sBh,
                   float* Cf, half_t* Ch, long ldc, long sCb, long sCh,
                   const float* __restrict__ bias,
                   int M, int N, int K, int H2) {
  __shared__ half_t As[2][128 * LROW];   // 2 x 10 KB
  __shared__ half_t Bs[2][64 * LROW];    // 2 x  5 KB

  const int z  = blockIdx.z;
  const int zb = z / H2, zh = z % H2;
  A  += (long)zb * sAb + (long)zh * sAh;
  Bt += (long)zb * sBb + (long)zh * sBh;
  const long coff = (long)zb * sCb + (long)zh * sCh;
  if (Cf) Cf += coff;
  if (Ch) Ch += coff;

  const int t    = threadIdx.x;
  const int lane = t & 31;
  const int wave = t >> 5;
  const int wm = wave & 3;          // 4 waves along M
  const int wn = wave >> 2;         // 2 waves along N
  const int mB = blockIdx.x * 128;
  const int nB = blockIdx.y * 64;
  const int m0 = wm * 32;           // wave tile origin within block tile
  const int n0 = wn * 32;
  const int laneLo = lane & 15;
  const int laneHi = (lane >> 4) & 1;

  // ---- async tile copy mapping: 16B chunks, 3 per thread per stage --------
  // A tile: 128 rows x 32 halves = 512 chunks (2/thread); B tile: 256 (1/thread)
  const int crow = t >> 2;            // 0..63
  const int ch   = (t & 3) * 8;       // half offset of chunk within row
  const int ar0  = imin(mB + crow,      M - 1);
  const int ar1  = imin(mB + 64 + crow, M - 1);
  const int brr  = imin(nB + crow,      N - 1);
  const half_t* gA0 = A  + (long)ar0 * lda + ch;
  const half_t* gA1 = A  + (long)ar1 * lda + ch;
  const half_t* gB  = Bt + (long)brr * ldb + ch;
  half_t* lA0[2] = { &As[0][crow * LROW + ch],        &As[1][crow * LROW + ch] };
  half_t* lA1[2] = { &As[0][(64 + crow) * LROW + ch], &As[1][(64 + crow) * LROW + ch] };
  half_t* lB [2] = { &Bs[0][crow * LROW + ch],        &Bs[1][crow * LROW + ch] };

  // ---- fragment LDS addresses (per stage) ---------------------------------
  // A frag (16x32): lane<16 K{0..7,16..23}; lane>=16 K{8..15,24..31}
  // B frag (32x16): lane<16 K{0..7,16..23}; lane>=16 K{8..15,24..31}
  const int aoff0 = (m0 + laneLo) * LROW + laneHi * 8;
  const int aoff1 = (m0 + 16 + laneLo) * LROW + laneHi * 8;
  const int boff0 = (n0 + laneLo) * LROW + laneHi * 8;
  const int boff1 = (n0 + 16 + laneLo) * LROW + laneHi * 8;

  v8f acc00 = {}, acc01 = {}, acc10 = {}, acc11 = {};

  // prologue: stage 0
  async_copy16(gA0, lA0[0]);
  async_copy16(gA1, lA1[0]);
  async_copy16(gB,  lB[0]);

  int s = 0;
  for (int k = 0; k < K; k += KSTEP) {
    const bool more = (k + KSTEP) < K;
    if (more) {
      async_copy16(gA0 + k + KSTEP, lA0[s ^ 1]);
      async_copy16(gA1 + k + KSTEP, lA1[s ^ 1]);
      async_copy16(gB  + k + KSTEP, lB [s ^ 1]);
      wait_async3();                 // stage s complete (3 newer in flight)
    } else {
      wait_async0();
    }
    __syncthreads();

    const half_t* As_ = As[s];
    const half_t* Bs_ = Bs[s];
    v16h a0 = load_frag2(As_ + aoff0,      As_ + aoff0 + 16);
    v16h a1 = load_frag2(As_ + aoff1,      As_ + aoff1 + 16);
    v16h b0 = load_frag2(Bs_ + boff0,      Bs_ + boff0 + 16);
    v16h b1 = load_frag2(Bs_ + boff1,      Bs_ + boff1 + 16);
    acc00 = __builtin_amdgcn_wmma_f32_16x16x32_f16(false, a0, false, b0,
                                                   (short)0, acc00, false, false);
    acc01 = __builtin_amdgcn_wmma_f32_16x16x32_f16(false, a0, false, b1,
                                                   (short)0, acc01, false, false);
    acc10 = __builtin_amdgcn_wmma_f32_16x16x32_f16(false, a1, false, b0,
                                                   (short)0, acc10, false, false);
    acc11 = __builtin_amdgcn_wmma_f32_16x16x32_f16(false, a1, false, b1,
                                                   (short)0, acc11, false, false);
    __syncthreads();                 // done reading stage s before overwrite
    s ^= 1;
  }

  // C/D layout: lane<16 -> N=laneLo, M=r; lane>=16 -> N=laneLo, M=8+r
  const int col0 = nB + n0      + laneLo;
  const int col1 = nB + n0 + 16 + laneLo;
  const int row0 = mB + m0      + laneHi * 8;
  const int row1 = mB + m0 + 16 + laneHi * 8;
  store_tile(acc00, row0, col0, M, N, ldc, Cf, Ch, bias);
  store_tile(acc01, row0, col1, M, N, ldc, Cf, Ch, bias);
  store_tile(acc10, row1, col0, M, N, ldc, Cf, Ch, bias);
  store_tile(acc11, row1, col1, M, N, ldc, Cf, Ch, bias);
}

// ---------------------------------------------------------------------------
// f32 -> f16 convert (4 elems/thread, 16B loads / 8B stores)
// ---------------------------------------------------------------------------
__global__ __launch_bounds__(256)
void cvt_f32_f16(const float* __restrict__ in, half_t* __restrict__ out, long n) {
  long i = ((long)blockIdx.x * blockDim.x + threadIdx.x) * 4;
  if (i >= n) return;
  float4 f = *(const float4*)(in + i);
  union { half_t h[4]; uint2 u; } o;
  o.h[0] = (half_t)f.x; o.h[1] = (half_t)f.y;
  o.h[2] = (half_t)f.z; o.h[3] = (half_t)f.w;
  *(uint2*)(out + i) = o.u;
}

// ---------------------------------------------------------------------------
// Tiled transpose + convert: in[R][C] f32 -> out[C][R] f16. Block (32,8).
// ---------------------------------------------------------------------------
__global__ __launch_bounds__(256)
void transpose_cvt(const float* __restrict__ in, half_t* __restrict__ out,
                   int R, int C) {
  __shared__ float tile[32][33];
  const int cx = blockIdx.x * 32, ry = blockIdx.y * 32;
  const int tx = threadIdx.x, ty = threadIdx.y;
#pragma unroll
  for (int i = 0; i < 32; i += 8)
    tile[ty + i][tx] = in[(long)(ry + ty + i) * C + (cx + tx)];
  __syncthreads();
#pragma unroll
  for (int i = 0; i < 32; i += 8)
    out[(long)(cx + ty + i) * R + (ry + tx)] = (half_t)tile[tx][ty + i];
}

// ---------------------------------------------------------------------------
// Per-(b,h): online softmax over sequence dim of k, then LDS-tiled transpose
// of normalized k and of v into Kt[d][n], Vt[e][n] (n contiguous, f16).
// Block = 256 threads, grid = B*H.
// ---------------------------------------------------------------------------
__global__ __launch_bounds__(256)
void softmax_kv_transpose(const half_t* __restrict__ qkv,
                          half_t* __restrict__ Kt,
                          half_t* __restrict__ Vt) {
  const int bh = blockIdx.x;
  const int b = bh >> 4, h = bh & 15;
  const half_t* kbase = qkv + (long)b * Nn * D3 + Dd     + h * HD;
  const half_t* vbase = qkv + (long)b * Nn * D3 + 2 * Dd + h * HD;
  const int t  = threadIdx.x;
  const int d  = t & 63;    // head-dim column
  const int nl = t >> 6;    // 0..3 row-phase

  __shared__ float sm[4][64], ss[4][64], colM[64], colS[64];
  __shared__ half_t ktile[64][72];   // [d][n_local], 16B-aligned rows
  __shared__ half_t vtile[64][72];

  // Pass 1: online (max, sum-exp) over N per column d
  float m = -3.0e38f, s = 0.0f;
  for (int n = nl; n < Nn; n += 4) {
    float xv = (float)kbase[(long)n * D3 + d];
    if (xv > m) { s = s * __expf(m - xv) + 1.0f; m = xv; }
    else        { s += __expf(xv - m); }
  }
  sm[nl][d] = m; ss[nl][d] = s;
  __syncthreads();
  if (t < 64) {
    float mm = sm[0][t], sa = ss[0][t];
#pragma unroll
    for (int i = 1; i < 4; ++i) {
      float m2 = sm[i][t], s2 = ss[i][t];
      float mn = fmaxf(mm, m2);
      sa = sa * __expf(mm - mn) + s2 * __expf(m2 - mn);
      mm = mn;
    }
    colM[t] = mm; colS[t] = 1.0f / sa;
  }
  __syncthreads();
  const float cm = colM[d], cs = colS[d];

  // Pass 2: 64x64 tiles; coalesced read along d, transposed write along n
  for (int tile = 0; tile < Nn / 64; ++tile) {
    const int nb = tile * 64;
#pragma unroll
    for (int i = 0; i < 16; ++i) {
      const int n = nb + nl * 16 + i;
      float kv = (float)kbase[(long)n * D3 + d];
      float vv = (float)vbase[(long)n * D3 + d];
      ktile[d][nl * 16 + i] = (half_t)(__expf(kv - cm) * cs);
      vtile[d][nl * 16 + i] = (half_t)vv;
    }
    __syncthreads();
    const int dd = t >> 2, j = t & 3;
    const uint4* kr = (const uint4*)&ktile[dd][j * 16];
    const uint4* vr = (const uint4*)&vtile[dd][j * 16];
    uint4 k0 = kr[0], k1 = kr[1], v0 = vr[0], v1 = vr[1];
    half_t* kdst = Kt + ((long)bh * HD + dd) * (long)Nn + nb + j * 16;
    half_t* vdst = Vt + ((long)bh * HD + dd) * (long)Nn + nb + j * 16;
    ((uint4*)kdst)[0] = k0; ((uint4*)kdst)[1] = k1;
    ((uint4*)vdst)[0] = v0; ((uint4*)vdst)[1] = v1;
    __syncthreads();
  }
}

// ---------------------------------------------------------------------------
// Launcher
// ---------------------------------------------------------------------------
extern "C" void kernel_launch(void* const* d_in, const int* in_sizes, int n_in,
                              void* d_out, int out_size, void* d_ws, size_t ws_size,
                              hipStream_t stream) {
  (void)in_sizes; (void)n_in; (void)out_size; (void)ws_size;
  const float* x    = (const float*)d_in[0];
  const float* Wqkv = (const float*)d_in[1];
  const float* Wout = (const float*)d_in[2];
  const float* bout = (const float*)d_in[3];
  float* out = (float*)d_out;

  char* ws = (char*)d_ws;
  // workspace layout (bytes)
  half_t* X16    = (half_t*)(ws);                         //  32 MB  x in f16
  half_t* WqkvT  = (half_t*)(ws + 33554432L);             //   6 MB  [3072][1024]
  half_t* WoutT  = (half_t*)(ws + 39845888L);             //   2 MB  [1024][1024]
  half_t* QKV16  = (half_t*)(ws + 41943040L);             //  96 MB  [16384][3072]
  half_t* Kt     = (half_t*)(ws + 142606336L);            //  32 MB  [64][64][4096]
  half_t* Vt     = (half_t*)(ws + 176160768L);            //  32 MB
  half_t* CTX16  = (half_t*)(ws + 209715200L);            // 0.5 MB  [64][64][64]
  half_t* ATTN16 = (half_t*)(ws + 210239488L);            //  32 MB  [16384][1024]

  // 1) x -> f16
  cvt_f32_f16<<<dim3((unsigned)((BN * Dd) / 4 / 256)), dim3(256), 0, stream>>>(
      x, X16, BN * Dd);
  // 2) weights -> transposed f16
  transpose_cvt<<<dim3(D3 / 32, Dd / 32), dim3(32, 8), 0, stream>>>(
      Wqkv, WqkvT, Dd, D3);
  transpose_cvt<<<dim3(Dd / 32, Dd / 32), dim3(32, 8), 0, stream>>>(
      Wout, WoutT, Dd, Dd);

  // 3) QKV16 = X16 x WqkvT   (M=16384, N=3072, K=1024)
  gemm_wmma_f16<<<dim3(128, 48, 1), dim3(256), 0, stream>>>(
      X16, Dd, 0, 0,  WqkvT, Dd, 0, 0,
      nullptr, QKV16, D3, 0, 0, nullptr,
      (int)BN, D3, Dd, 1);

  // 4) softmax over N on k, transpose k and v per head
  softmax_kv_transpose<<<dim3(Bb * Hh), dim3(256), 0, stream>>>(QKV16, Kt, Vt);

  // 5) CTX16[e][d] = Vt x Kt^T per (b,h)  (M=64, N=64, K=4096, batch 64)
  gemm_wmma_f16<<<dim3(1, 1, Bb * Hh), dim3(256), 0, stream>>>(
      Vt, Nn, (long)Hh * HD * Nn, (long)HD * Nn,
      Kt, Nn, (long)Hh * HD * Nn, (long)HD * Nn,
      nullptr, CTX16, HD, (long)Hh * HD * HD, (long)HD * HD, nullptr,
      HD, HD, Nn, Hh);

  // 6) ATTN16[n][h*64+e] = q x ctx per (b,h)  (M=4096, N=64, K=64, batch 64)
  gemm_wmma_f16<<<dim3(32, 1, Bb * Hh), dim3(256), 0, stream>>>(
      QKV16, D3, (long)Nn * D3, (long)HD,
      CTX16, HD, (long)Hh * HD * HD, (long)HD * HD,
      nullptr, ATTN16, Dd, (long)Nn * Dd, (long)HD, nullptr,
      Nn, HD, HD, Hh);

  // 7) out = ATTN16 x WoutT + b_out  (M=16384, N=1024, K=1024, f32 out)
  gemm_wmma_f16<<<dim3(128, 16, 1), dim3(256), 0, stream>>>(
      ATTN16, Dd, 0, 0,  WoutT, Dd, 0, 0,
      out, nullptr, Dd, 0, 0, bout,
      (int)BN, Dd, Dd, 1);
}
